// Attention_15925738733878
// MI455X (gfx1250) — matrix operand order repaired and checked by reading, not compiled, not run on hardware
//
#include <hip/hip_runtime.h>

typedef __attribute__((ext_vector_type(16))) _Float16 v16h;
typedef __attribute__((ext_vector_type(8)))  float    v8f;

#define B_    64
#define H_    32
#define KV_   8
#define HD_   128
#define BS_   16
#define MB_   128
#define GQ_   4
#define SCALE_ 0.08838834764831845f

#define NWAVE   4
#define VSTRIDE 128   // V-tile row stride (floats); transposed reads are lane-contiguous
#define PSTRIDE 16    // P-tile row stride (halfs): 32B rows -> aligned ds_load_b128

#if __has_builtin(__builtin_amdgcn_global_load_async_to_lds_b128)
#define HAS_ASYNC_LDS 1
// Builtin signature (from the round-2 diagnostic): param0 is a non-const
// 'int __vector(4) addrspace(1)*' (global src), param1 the LDS dst, then
// imm offset and imm cpol.
typedef int v4i_ __attribute__((vector_size(16)));
typedef __attribute__((address_space(1))) v4i_* gv4p_;
typedef __attribute__((address_space(3))) v4i_* lv4p_;
#define ASYNC_CP16(dst, src)                                             \
  __builtin_amdgcn_global_load_async_to_lds_b128((gv4p_)(src),           \
                                                 (lv4p_)(dst), 0, 0)
#else
#define HAS_ASYNC_LDS 0
#endif

static __device__ __forceinline__ void wait_async0() {
#if HAS_ASYNC_LDS
#if __has_builtin(__builtin_amdgcn_s_wait_asynccnt)
  __builtin_amdgcn_s_wait_asynccnt(0);
#else
  asm volatile("s_wait_asynccnt 0x0" ::: "memory");
#endif
#endif
}

__launch_bounds__(128, 1)
__global__ void paged_decode_wmma(
    const float* __restrict__ q,
    const float* __restrict__ knew,
    const float* __restrict__ vnew,
    const float* __restrict__ kcache,
    const float* __restrict__ vcache,
    const int*   __restrict__ block_tables,
    const int*   __restrict__ context_lens,
    float* __restrict__ out)
{
  __shared__ float    Vt[NWAVE][BS_ * VSTRIDE];
  __shared__ _Float16 Pt[NWAVE][16 * PSTRIDE];
  __shared__ float    Opart[NWAVE][GQ_][HD_];
  __shared__ float    Wm[NWAVE][GQ_];
  __shared__ float    Wl[NWAVE][GQ_];

  const int bkv = blockIdx.x;
  const int b   = bkv / KV_;
  const int kv  = bkv % KV_;
  const int tid = threadIdx.x;
  // wave id is wave-uniform: force it into an SGPR so the whole block/pointer
  // arithmetic in the loop is scalar (s_load for block_tables, s_cmp for fresh).
  const int w   = __builtin_amdgcn_readfirstlane(tid >> 5);
  const int l   = tid & 31;   // lane
  const int lm  = l & 15;     // row (A frags) or column/token (B/C frags)
  const int lh  = l >> 4;     // lane-half selector in fragment layouts

  const int ctx    = context_lens[b];
  const int lastib = (ctx - 1) >> 4;          // only block holding the fresh token
  const int nblk   = (ctx + BS_ - 1) / BS_;

  // zero the P tile once: rows 4..15 stay zero forever -> padded WMMA rows exact zeros
  for (int i = l; i < 16 * PSTRIDE; i += 32) Pt[w][i] = (_Float16)0.0f;

  // ---------------- Q fragments (A-matrix 16x32 f16), rows >= 4 are zero ----------------
  v16h qf[4];
  #pragma unroll
  for (int kc = 0; kc < 4; ++kc)
    #pragma unroll
    for (int e = 0; e < 16; ++e) qf[kc][e] = (_Float16)0.0f;
  if (lm < GQ_) {
    const float* qrow = q + ((size_t)b * H_ + kv * GQ_ + lm) * HD_;
    #pragma unroll
    for (int kc = 0; kc < 4; ++kc) {
      #pragma unroll
      for (int e = 0; e < 16; ++e) {
        const int kk = 32 * kc + 16 * lh + e;
        qf[kc][e] = (_Float16)(qrow[kk] * SCALE_);
      }
    }
  }

  const float* kfresh = knew + ((size_t)b * KV_ + kv) * HD_;
  const float* vfresh = vnew + ((size_t)b * KV_ + kv) * HD_;

  v8f oacc[8];
  #pragma unroll
  for (int ht = 0; ht < 8; ++ht)
    #pragma unroll
    for (int r = 0; r < 8; ++r) oacc[ht][r] = 0.0f;

  float m_run[GQ_], l_run[GQ_];
  #pragma unroll
  for (int r = 0; r < GQ_; ++r) { m_run[r] = -1e30f; l_run[r] = 0.0f; }

  for (int ib = w; ib < nblk; ib += NWAVE) {
    const int t0  = ib * BS_;
    const int blk = block_tables[b * MB_ + ib];            // scalar load
    const float* kblock = kcache + ((size_t)blk * BS_) * (KV_ * HD_) + (size_t)kv * HD_;
    const float* vblock = vcache + ((size_t)blk * BS_) * (KV_ * HD_) + (size_t)kv * HD_;
    const bool fresh = (ib == lastib);                     // scalar branch

    // ---- V tile -> LDS: async DMA (no VGPR round trip), one b128/lane per row ----
#if HAS_ASYNC_LDS
    if (!fresh) {
      #pragma unroll
      for (int i = 0; i < BS_; ++i)
        ASYNC_CP16(&Vt[w][i * VSTRIDE + 4 * l],
                   vblock + (size_t)i * (KV_ * HD_) + 4 * l);
    } else {
      #pragma unroll
      for (int i = 0; i < BS_; ++i) {
        const float* vrow = (t0 + i == ctx - 1) ? vfresh
                                                : (vblock + (size_t)i * (KV_ * HD_));
        ASYNC_CP16(&Vt[w][i * VSTRIDE + 4 * l], vrow + 4 * l);
      }
    }
#else
    #pragma unroll
    for (int i = 0; i < BS_; ++i) {
      const float* vrow = (fresh && (t0 + i == ctx - 1))
                              ? vfresh
                              : (vblock + (size_t)i * (KV_ * HD_));
      const float4 d = ((const float4*)vrow)[l];
      *((float4*)&Vt[w][i * VSTRIDE + 4 * l]) = d;
    }
#endif

    // ---- prefetch next block's K and V lines (HBM lead time ~1 block) ----
    const int ibn = ib + NWAVE;
    if (ibn < nblk) {
      const int blkn = block_tables[b * MB_ + ibn];
      const float* kn = kcache + ((size_t)blkn * BS_) * (KV_ * HD_) + (size_t)kv * HD_;
      const float* vn = vcache + ((size_t)blkn * BS_) * (KV_ * HD_) + (size_t)kv * HD_;
      const size_t ro = (size_t)lm * (KV_ * HD_) + 32 * lh;
      __builtin_prefetch(kn + ro, 0, 3);
      __builtin_prefetch(kn + ro + 64, 0, 3);
      __builtin_prefetch(vn + ro, 0, 3);
      __builtin_prefetch(vn + ro + 64, 0, 3);
    }

    // ---- K row base for B-fragment (lane = token) ----
    const float* krow;
    if (!fresh) {
      krow = kblock + (size_t)lm * (KV_ * HD_);
    } else {
      krow = (t0 + lm == ctx - 1) ? kfresh : (kblock + (size_t)lm * (KV_ * HD_));
    }

    // ---- scores = Q * K^T : 4 WMMAs over HD=128 ----
    v8f sc;
    #pragma unroll
    for (int r = 0; r < 8; ++r) sc[r] = 0.0f;
    #pragma unroll
    for (int kc = 0; kc < 4; ++kc) {
      v16h kb;
      const float* kseg = krow + 32 * kc + 16 * lh;   // 16 contiguous floats -> 4x b128
      #pragma unroll
      for (int e = 0; e < 16; ++e) kb[e] = (_Float16)kseg[e];
      sc = __builtin_amdgcn_wmma_f32_16x16x32_f16(false, qf[kc], false, kb,
                                                  (short)0, sc, false, false);
    }

    // ---- online softmax over this 16-token tile (rows 0..3 live in lanes 0..15) ----
    const bool valid = (t0 + lm) < ctx;
    float fac[GQ_], pv[GQ_];
    #pragma unroll
    for (int r = 0; r < GQ_; ++r) {
      float s  = valid ? sc[r] : -1e30f;
      float mx = s;
      #pragma unroll
      for (int d = 1; d < 16; d <<= 1) mx = fmaxf(mx, __shfl_xor(mx, d, 32));
      const float mn = fmaxf(m_run[r], mx);
      fac[r]   = __expf(m_run[r] - mn);
      m_run[r] = mn;
      const float p = __expf(s - mn);
      pv[r] = p;
      float sum = p;
      #pragma unroll
      for (int d = 1; d < 16; d <<= 1) sum += __shfl_xor(sum, d, 32);
      l_run[r] = l_run[r] * fac[r] + sum;
    }

    // store P rows 0..3 as f16 (K dim 16..31 of the PV WMMA stays zero)
    if (l < 16) {
      #pragma unroll
      for (int r = 0; r < GQ_; ++r) Pt[w][r * PSTRIDE + l] = (_Float16)pv[r];
    }

    // rescale output accumulators
    #pragma unroll
    for (int ht = 0; ht < 8; ++ht)
      #pragma unroll
      for (int r = 0; r < GQ_; ++r) oacc[ht][r] *= fac[r];

    // ---- P A-fragment from LDS: rows 32B apart -> two aligned ds_load_b128 ----
    v16h pa;
    #pragma unroll
    for (int e = 0; e < 16; ++e) pa[e] = (_Float16)0.0f;
    #pragma unroll
    for (int e = 0; e < 16; ++e) if (lh == 0) pa[e] = Pt[w][lm * PSTRIDE + e];

    wait_async0();   // V tile resident in LDS before the transposed gather

    // ---- O += P * V : 8 WMMAs across HD tiles ----
    #pragma unroll
    for (int ht = 0; ht < 8; ++ht) {
      v16h vb;
      if (lh == 0) {
        #pragma unroll
        for (int e = 0; e < 16; ++e)
          vb[e] = (_Float16)Vt[w][e * VSTRIDE + 16 * ht + lm];
      } else {
        #pragma unroll
        for (int e = 0; e < 16; ++e) vb[e] = (_Float16)0.0f;  // tokens 16..31 (padding)
      }
      oacc[ht] = __builtin_amdgcn_wmma_f32_16x16x32_f16(false, pa, false, vb,
                                                        (short)0, oacc[ht], false, false);
    }
  }

  // ---------------- publish per-wave partials ----------------
  if (l == 0) {
    #pragma unroll
    for (int r = 0; r < GQ_; ++r) { Wm[w][r] = m_run[r]; Wl[w][r] = l_run[r]; }
  }
  if (l < 16) {
    #pragma unroll
    for (int ht = 0; ht < 8; ++ht)
      #pragma unroll
      for (int r = 0; r < GQ_; ++r)
        Opart[w][r][16 * ht + l] = oacc[ht][r];
  }
  __syncthreads();

  // ---------------- cross-wave log-sum-exp combine + store ----------------
  const int hd = tid & (HD_ - 1);
  #pragma unroll
  for (int m = 0; m < GQ_; ++m) {
    float M = Wm[0][m];
    #pragma unroll
    for (int ww = 1; ww < NWAVE; ++ww) M = fmaxf(M, Wm[ww][m]);
    float L = 0.0f, val = 0.0f;
    #pragma unroll
    for (int ww = 0; ww < NWAVE; ++ww) {
      const float f = __expf(Wm[ww][m] - M);
      L   += Wl[ww][m] * f;
      val += Opart[ww][m][hd] * f;
    }
    out[((size_t)b * H_ + kv * GQ_ + m) * HD_ + hd] = val / L;
  }
}

extern "C" void kernel_launch(void* const* d_in, const int* in_sizes, int n_in,
                              void* d_out, int out_size, void* d_ws, size_t ws_size,
                              hipStream_t stream) {
  (void)in_sizes; (void)n_in; (void)out_size; (void)d_ws; (void)ws_size;
  const float* q  = (const float*)d_in[0];
  const float* k  = (const float*)d_in[1];
  const float* v  = (const float*)d_in[2];
  const float* kc = (const float*)d_in[3];
  const float* vc = (const float*)d_in[4];
  // d_in[5] (slot_mapping) not needed: fresh k/v substituted at t == ctx-1,
  // exactly where the reference scatters them (blocks are sequence-private).
  const int* bt = (const int*)d_in[6];
  const int* cl = (const int*)d_in[7];
  paged_decode_wmma<<<dim3(B_ * KV_), dim3(128), 0, stream>>>(
      q, k, v, kc, vc, bt, cl, (float*)d_out);
}
